// CapsuleNet_19370302505577
// MI455X (gfx1250) — matrix operand order repaired and checked by reading, not compiled, not run on hardware
//
#include <hip/hip_runtime.h>
#include <hip/hip_bf16.h>

// ---------------- problem constants ----------------
#define BB   64     // batch
#define PP   24     // acoustic params
#define CC   40     // classes
#define WINW 64     // window
#define L1L  60     // after conv1 (k=5)
#define L2L  56     // after prim conv (k=5)
#define RR   1344   // routes = PP*L2L
#define ICL  32     // capsule in-dim
#define OCL  16     // capsule out-dim

#define NGEMM (BB * L2L)   // 3584
#define KGEMM 1280         // 256*5
#define MGEMM 768          // PP*32

typedef __attribute__((ext_vector_type(16))) _Float16 v16h;
typedef __attribute__((ext_vector_type(8)))  _Float16 v8h;
typedef __attribute__((ext_vector_type(8)))  float    v8f;

union FragH16 { v16h v; v8h h2[2]; _Float16 e[16]; };
union Out8H   { v8h  v; _Float16 e[8]; };

// CDNA5 async global->LDS copy (GLOBAL_LOAD_ASYNC_TO_LDS_B128, ASYNCcnt).
// vdst = per-lane LDS byte offset (low 32 bits of a generic LDS pointer),
// vaddr = 64-bit global address; imm offset applies to both sides.
__device__ __forceinline__ void async_copy_b128(const void* gsrc, void* lds_dst) {
    unsigned           ldsa = (unsigned)(unsigned long long)lds_dst;
    unsigned long long ga   = (unsigned long long)gsrc;
    asm volatile("global_load_async_to_lds_b128 %0, %1, off"
                 :: "v"(ldsa), "v"(ga) : "memory");
}
__device__ __forceinline__ void async_copy_b128_off16(const void* gsrc, void* lds_dst) {
    unsigned           ldsa = (unsigned)(unsigned long long)lds_dst;
    unsigned long long ga   = (unsigned long long)gsrc;
    asm volatile("global_load_async_to_lds_b128 %0, %1, off\n\t"
                 "global_load_async_to_lds_b128 %0, %1, off offset:16"
                 :: "v"(ldsa), "v"(ga) : "memory");
}
__device__ __forceinline__ void wait_async0() {
    asm volatile("s_wait_asynccnt 0x0" ::: "memory");
}

// ---------------- utility kernels ----------------
__global__ void zero_f32_kernel(float* p, int n) {
    int i = blockIdx.x * blockDim.x + threadIdx.x;
    if (i < n) p[i] = 0.0f;
}

__global__ void f32_to_f16_kernel(const float* src, _Float16* dst, int n) {
    int i = blockIdx.x * blockDim.x + threadIdx.x;
    if (i < n) dst[i] = (_Float16)src[i];
}

// ---------------- stage 1: conv1 (24->256, k=5) ----------------
// x: (B, WIN, P) ; w: (256, 24, 5) ; out: conv_raw[(b*256+oc)*60 + l]
__global__ void conv1_kernel(const float* __restrict__ x,
                             const float* __restrict__ w,
                             const float* __restrict__ bias,
                             float* __restrict__ out) {
    __shared__ float xs[120];                 // 5 rows x 24 params
    const int b = blockIdx.x / L1L;
    const int l = blockIdx.x % L1L;
    const int t = threadIdx.x;
    if (t < 120) {
        int k  = t / PP;
        int ic = t % PP;
        xs[t] = x[((size_t)b * WINW + (l + k)) * PP + ic];
    }
    __syncthreads();
    const int oc = t;                          // 256 threads
    float acc = bias[oc];
    const float* wr = w + (size_t)oc * 120;
    #pragma unroll
    for (int ic = 0; ic < PP; ++ic)
        #pragma unroll
        for (int k = 0; k < 5; ++k)
            acc += xs[k * PP + ic] * wr[ic * 5 + k];
    out[((size_t)b * 256 + oc) * L1L + l] = acc;
}

// BN1 stats: per channel over (B, L1)  => stat[oc]=mean, stat[256+oc]=inv_std
__global__ void stats1_kernel(const float* __restrict__ conv_raw, float* __restrict__ stat) {
    __shared__ float rs[256], rs2[256];
    const int oc = blockIdx.x;
    const int t  = threadIdx.x;
    float s = 0.f, s2 = 0.f;
    for (int idx = t; idx < BB * L1L; idx += 256) {
        int b = idx / L1L, l = idx % L1L;
        float v = conv_raw[((size_t)b * 256 + oc) * L1L + l];
        s += v; s2 += v * v;
    }
    rs[t] = s; rs2[t] = s2;
    __syncthreads();
    for (int off = 128; off > 0; off >>= 1) {
        if (t < off) { rs[t] += rs[t + off]; rs2[t] += rs2[t + off]; }
        __syncthreads();
    }
    if (t == 0) {
        float inv_n = 1.0f / (BB * L1L);
        float m = rs[0] * inv_n;
        float v = rs2[0] * inv_n - m * m;
        stat[oc]       = m;
        stat[256 + oc] = rsqrtf(v + 1e-5f);
    }
}

// BN1 apply + ReLU + im2col -> hcol[n=b*56+l][k=ic*5+kk] (f16), K=1280
__global__ void im2col_bn1_kernel(const float* __restrict__ conv_raw,
                                  const float* __restrict__ stat,
                                  const float* __restrict__ g,
                                  const float* __restrict__ bbn,
                                  _Float16* __restrict__ hcol) {
    const int n = blockIdx.x;
    const int b = n / L2L, l = n % L2L;
    for (int k = threadIdx.x; k < KGEMM; k += 256) {
        int ic = k / 5, kk = k % 5;
        float v = conv_raw[((size_t)b * 256 + ic) * L1L + (l + kk)];
        v = g[ic] * (v - stat[ic]) * stat[256 + ic] + bbn[ic];
        v = fmaxf(v, 0.0f);
        hcol[(size_t)n * KGEMM + k] = (_Float16)v;
    }
}

// ---------------- stage 2: primary-capsule conv as WMMA GEMM ----------------
// C(768 x 3584) = Aw(768 x 1280) * hcol^T ;  u_raw[m*3584+n] = C + bias[m]
__global__ void gemm_prim_kernel(const _Float16* __restrict__ Aw,
                                 const _Float16* __restrict__ Bh,
                                 const float* __restrict__ bias,
                                 float* __restrict__ Cout) {
    __shared__ alignas(32) _Float16 ldsA[128 * 32];
    __shared__ alignas(32) _Float16 ldsB[64 * 32];
    const int t    = threadIdx.x;
    const int lane = t & 31;
    const int wid  = t >> 5;           // 8 waves
    const int wm   = wid >> 1;         // 0..3 (M)
    const int wn   = wid & 1;          // 0..1 (N)
    const int h    = lane >> 4;        // half-wave
    const int row  = lane & 15;
    const int mBase = blockIdx.x * 128;
    const int nBase = blockIdx.y * 64;

    v8f acc[2][2] = {};

    for (int kt = 0; kt < KGEMM / 32; ++kt) {
        const int k0 = kt * 32;
        // stage A tile 128x32 : 16 halfs / thread via async global->LDS (2x b128)
        {
            int m = t >> 1, part = t & 1;
            const _Float16* src = Aw + (size_t)(mBase + m) * KGEMM + k0 + part * 16;
            async_copy_b128_off16(src, &ldsA[m * 32 + part * 16]);
            if (kt + 1 < KGEMM / 32)
                __builtin_prefetch(src + 32, 0, 0);   // global_prefetch_b8 next K-tile
        }
        // stage B tile 64x32 : 8 halfs / thread via async global->LDS (1x b128)
        {
            int n = t >> 2, seg = t & 3;
            const _Float16* src = Bh + (size_t)(nBase + n) * KGEMM + k0 + seg * 8;
            async_copy_b128(src, &ldsB[n * 32 + seg * 8]);
        }
        wait_async0();          // ASYNCcnt -> 0 : LDS tiles written
        __syncthreads();

        FragH16 af[2]; v16h bf[2];
        #pragma unroll
        for (int sm = 0; sm < 2; ++sm) {
            int mrow = wm * 32 + sm * 16 + row;
            af[sm].v = *(const v16h*)(&ldsA[mrow * 32 + h * 16]);
        }
        #pragma unroll
        for (int sn = 0; sn < 2; ++sn) {
            int nrow = wn * 32 + sn * 16 + row;
            bf[sn] = *(const v16h*)(&ldsB[nrow * 32 + h * 16]);
        }
        #pragma unroll
        for (int sm = 0; sm < 2; ++sm)
            #pragma unroll
            for (int sn = 0; sn < 2; ++sn)
                acc[sm][sn] = __builtin_amdgcn_wmma_f32_16x16x32_f16(
                    false, af[sm].v, false, bf[sn], (short)0, acc[sm][sn], false, false);
        __syncthreads();
    }
    // epilogue: D layout M = v + h*8, N = row
    #pragma unroll
    for (int sm = 0; sm < 2; ++sm)
        #pragma unroll
        for (int sn = 0; sn < 2; ++sn) {
            int n = nBase + wn * 32 + sn * 16 + row;
            #pragma unroll
            for (int v = 0; v < 8; ++v) {
                int m = mBase + wm * 32 + sm * 16 + v + h * 8;
                Cout[(size_t)m * NGEMM + n] = acc[sm][sn][v] + bias[m];
            }
        }
}

// BN2 stats over (b, p, l) per capsule channel j<32
__global__ void stats2_kernel(const float* __restrict__ u_raw, float* __restrict__ stat) {
    __shared__ float rs[256], rs2[256];
    const int j = blockIdx.x;
    const int t = threadIdx.x;
    float s = 0.f, s2 = 0.f;
    for (int idx = t; idx < PP * NGEMM; idx += 256) {
        int p = idx / NGEMM, n = idx % NGEMM;
        float v = u_raw[((size_t)(p * 32 + j)) * NGEMM + n];
        s += v; s2 += v * v;
    }
    rs[t] = s; rs2[t] = s2;
    __syncthreads();
    for (int off = 128; off > 0; off >>= 1) {
        if (t < off) { rs[t] += rs[t + off]; rs2[t] += rs2[t + off]; }
        __syncthreads();
    }
    if (t == 0) {
        float inv_n = 1.0f / (PP * NGEMM);
        float m = rs[0] * inv_n;
        float v = rs2[0] * inv_n - m * m;
        stat[j]      = m;
        stat[32 + j] = rsqrtf(v + 1e-5f);
    }
}

// BN2 apply + squash over 32-dim; write u_f16[(r*64 + b)*32 + i]
__global__ void bn_squash_kernel(const float* __restrict__ u_raw,
                                 const float* __restrict__ stat,
                                 const float* __restrict__ g2,
                                 const float* __restrict__ b2,
                                 _Float16* __restrict__ u16) {
    __shared__ float sd[256];
    const int t    = threadIdx.x;
    const int pair = blockIdx.x * 8 + (t >> 5);   // (b,r) pair per 32-lane group
    const int i    = t & 31;
    const int b = pair / RR;
    const int r = pair % RR;
    const int p = r / L2L, l = r % L2L;
    const int n = b * L2L + l;
    float x   = u_raw[((size_t)(p * 32 + i)) * NGEMM + n];
    float val = g2[i] * (x - stat[i]) * stat[32 + i] + b2[i];
    sd[t] = val * val;
    __syncthreads();
    float sq = 0.f;
    int base = (t >> 5) << 5;
    #pragma unroll
    for (int k = 0; k < 32; ++k) sq += sd[base + k];
    float out = val * (sq / (1.0f + sq)) / (sqrtf(sq) + 1e-8f);
    u16[((size_t)r * BB + b) * 32 + i] = (_Float16)out;
}

// ---------------- stage 3: u_hat einsum via WMMA ----------------
// u_hat[b,c,r,d] = sum_i Wc[c,r,d,i] * u[b,r,i]
// block = one route r, 8 waves x 5 classes each, 4 batch tiles of 16
__global__ void uhat_kernel(const _Float16* __restrict__ Wc,
                            const _Float16* __restrict__ U16,
                            _Float16* __restrict__ uhat) {
    __shared__ alignas(32) _Float16 ldsU[BB * 32];   // 4 KB
    const int r    = blockIdx.x;
    const int t    = threadIdx.x;
    const int lane = t & 31;
    const int wid  = t >> 5;
    const int h    = lane >> 4;
    const int mrow = lane & 15;
    // stage u[:, r, :] : 2048 halfs, 8 per thread (async global->LDS)
    async_copy_b128(U16 + (size_t)r * (BB * 32) + t * 8, &ldsU[t * 8]);
    wait_async0();
    __syncthreads();

    v16h bf[4];
    #pragma unroll
    for (int nt = 0; nt < 4; ++nt) {
        int bl = nt * 16 + mrow;
        bf[nt] = *(const v16h*)(&ldsU[bl * 32 + h * 16]);
    }
    #pragma unroll
    for (int ccy = 0; ccy < 5; ++ccy) {
        const int c = wid + 8 * ccy;
        const _Float16* wrow = Wc + (((size_t)c * RR + r) * 16 + mrow) * 32;
        __builtin_prefetch(wrow + 16 * 32, 0, 0);
        FragH16 af;
        af.v = *(const v16h*)(wrow + h * 16);
        #pragma unroll
        for (int nt = 0; nt < 4; ++nt) {
            v8f acc = {};
            acc = __builtin_amdgcn_wmma_f32_16x16x32_f16(
                false, af.v, false, bf[nt], (short)0, acc, false, false);
            const int bo = nt * 16 + mrow;                 // N = batch
            Out8H o;
            #pragma unroll
            for (int v = 0; v < 8; ++v) o.e[v] = (_Float16)acc[v];
            _Float16* dst = uhat + (((size_t)bo * CC + c) * RR + r) * 16 + h * 8;
            *(v8h*)dst = o.v;                              // d = v + h*8
        }
    }
}

// ---------------- stage 4: dynamic routing ----------------
// b_ij layout: (B, R, C) ; c_ij same
__global__ void softmax_kernel(const float* __restrict__ bij, float* __restrict__ cij) {
    const int row = blockIdx.x * blockDim.x + threadIdx.x;   // b*R + r
    if (row >= BB * RR) return;
    const float* src = bij + (size_t)row * CC;
    float* dst = cij + (size_t)row * CC;
    float mx = src[0];
    for (int c = 1; c < CC; ++c) mx = fmaxf(mx, src[c]);
    float sum = 0.f;
    for (int c = 0; c < CC; ++c) sum += expf(src[c] - mx);
    float inv = 1.0f / sum;
    for (int c = 0; c < CC; ++c) dst[c] = expf(src[c] - mx) * inv;
}

// s_j[b,c,:] = sum_r c_ij * u_hat ; v = squash(s)
__global__ void route_s_kernel(const _Float16* __restrict__ uhat,
                               const float* __restrict__ cij,
                               float* __restrict__ vbuf) {
    __shared__ float red[256];
    const int bc = blockIdx.x;
    const int b = bc / CC, c = bc % CC;
    const int t = threadIdx.x;
    const int d = t & 15, rl = t >> 4;
    float acc = 0.f;
    for (int r = rl; r < RR; r += 16) {
        float w = cij[((size_t)b * RR + r) * CC + c];
        acc += w * (float)uhat[(((size_t)b * CC + c) * RR + r) * 16 + d];
    }
    red[t] = acc;
    __syncthreads();
    if (t < 16) {
        float s = 0.f;
        for (int g = 0; g < 16; ++g) s += red[g * 16 + t];
        red[t] = s;
    }
    __syncthreads();
    if (t < 16) {
        float sq = 0.f;
        for (int dd = 0; dd < 16; ++dd) sq += red[dd] * red[dd];
        float sv = red[t];
        vbuf[(size_t)bc * 16 + t] = sv * (sq / (1.0f + sq)) / (sqrtf(sq) + 1e-8f);
    }
}

// b_ij += <u_hat[b,c,r,:], v[b,c,:]>
__global__ void route_b_kernel(const _Float16* __restrict__ uhat,
                               const float* __restrict__ vbuf,
                               float* __restrict__ bij) {
    __shared__ float vv[16];
    const int bc = blockIdx.x;
    const int b = bc / CC, c = bc % CC;
    const int t = threadIdx.x;
    if (t < 16) vv[t] = vbuf[(size_t)bc * 16 + t];
    __syncthreads();
    for (int r = t; r < RR; r += 256) {
        const _Float16* row = uhat + (((size_t)b * CC + c) * RR + r) * 16;
        float dot = 0.f;
        #pragma unroll
        for (int d = 0; d < 16; ++d) dot += (float)row[d] * vv[d];
        bij[((size_t)b * RR + r) * CC + c] += dot;
    }
}

// ---------------- stage 5: classifier + decoder ----------------
__global__ void cls1_kernel(const float* __restrict__ vbuf, const float* __restrict__ w1,
                            const float* __restrict__ b1, float* __restrict__ hidden) {
    __shared__ float flat[OCL * CC];
    const int b = blockIdx.x, t = threadIdx.x;
    for (int k = t; k < OCL * CC; k += 256) flat[k] = vbuf[(size_t)b * OCL * CC + k];
    __syncthreads();
    float acc = b1[t];
    const float* wr = w1 + (size_t)t * (OCL * CC);
    for (int k = 0; k < OCL * CC; ++k) acc += flat[k] * wr[k];
    hidden[(size_t)b * 256 + t] = fmaxf(acc, 0.0f);
}

__global__ void cls2_kernel(const float* __restrict__ hidden, const float* __restrict__ w2,
                            const float* __restrict__ b2, float* __restrict__ dout,
                            int* __restrict__ amax) {
    __shared__ float hh[256];
    __shared__ float lg[CC];
    const int b = blockIdx.x, t = threadIdx.x;
    for (int k = t; k < 256; k += 64) hh[k] = hidden[(size_t)b * 256 + k];
    __syncthreads();
    if (t < CC) {
        float acc = b2[t];
        const float* wr = w2 + (size_t)t * 256;
        for (int k = 0; k < 256; ++k) acc += hh[k] * wr[k];
        lg[t] = acc;
    }
    __syncthreads();
    if (t == 0) {
        float mx = lg[0]; int am = 0;
        for (int c = 1; c < CC; ++c) if (lg[c] > mx) { mx = lg[c]; am = c; }
        float sum = 0.f;
        for (int c = 0; c < CC; ++c) sum += expf(lg[c] - mx);
        float inv = 1.0f / sum;
        for (int c = 0; c < CC; ++c)
            dout[(size_t)BB * WINW * PP + (size_t)b * CC + c] = expf(lg[c] - mx) * inv;
        amax[b] = am;
    }
}

__global__ void dec1_kernel(const float* __restrict__ vbuf, const int* __restrict__ amax,
                            const float* __restrict__ w1, const float* __restrict__ b1,
                            float* __restrict__ hidden2) {
    __shared__ float vv[16];
    __shared__ int a_s;
    const int b = blockIdx.x, t = threadIdx.x;
    if (t == 0) a_s = amax[b];
    __syncthreads();
    const int a = a_s;
    if (t < 16) vv[t] = vbuf[((size_t)b * CC + a) * 16 + t];
    __syncthreads();
    float acc = b1[t];
    const float* wr = w1 + (size_t)t * (OCL * CC) + a * 16;
    #pragma unroll
    for (int d = 0; d < 16; ++d) acc += vv[d] * wr[d];
    hidden2[(size_t)b * 256 + t] = fmaxf(acc, 0.0f);
}

__global__ void dec2_kernel(const float* __restrict__ hidden2, const float* __restrict__ w2,
                            const float* __restrict__ b2, float* __restrict__ dout) {
    __shared__ float hh[256];
    const int b = blockIdx.x, t = threadIdx.x;
    hh[t] = hidden2[(size_t)b * 256 + t];
    __syncthreads();
    for (int i = t; i < WINW * PP; i += 256) {
        float acc = b2[i];
        const float* wr = w2 + (size_t)i * 256;
        for (int k = 0; k < 256; ++k) acc += hh[k] * wr[k];
        dout[(size_t)b * (WINW * PP) + i] = 1.0f / (1.0f + expf(-acc));
    }
}

// ---------------- launcher ----------------
static inline size_t aln(size_t x) { return (x + 255) & ~(size_t)255; }

extern "C" void kernel_launch(void* const* d_in, const int* in_sizes, int n_in,
                              void* d_out, int out_size, void* d_ws, size_t ws_size,
                              hipStream_t stream) {
    const float* x       = (const float*)d_in[0];
    const float* conv1_w = (const float*)d_in[1];
    const float* conv1_b = (const float*)d_in[2];
    const float* bn1_g   = (const float*)d_in[3];
    const float* bn1_b   = (const float*)d_in[4];
    const float* prim_w  = (const float*)d_in[5];
    const float* prim_b  = (const float*)d_in[6];
    const float* bn2_g   = (const float*)d_in[7];
    const float* bn2_b   = (const float*)d_in[8];
    const float* W_cap   = (const float*)d_in[9];
    const float* cls_w1  = (const float*)d_in[10];
    const float* cls_b1  = (const float*)d_in[11];
    const float* cls_w2  = (const float*)d_in[12];
    const float* cls_b2  = (const float*)d_in[13];
    const float* dec_w1  = (const float*)d_in[14];
    const float* dec_b1  = (const float*)d_in[15];
    const float* dec_w2  = (const float*)d_in[16];
    const float* dec_b2  = (const float*)d_in[17];
    float* dout = (float*)d_out;

    char* ws = (char*)d_ws;
    size_t off = 0;
    float*    conv_raw = (float*)(ws + off);    off = aln(off + (size_t)BB * 256 * L1L * 4);
    float*    stat1    = (float*)(ws + off);    off = aln(off + 512 * 4);
    _Float16* hcol     = (_Float16*)(ws + off); off = aln(off + (size_t)NGEMM * KGEMM * 2);
    _Float16* wprim16  = (_Float16*)(ws + off); off = aln(off + (size_t)MGEMM * KGEMM * 2);
    float*    u_raw    = (float*)(ws + off);    off = aln(off + (size_t)MGEMM * NGEMM * 4);
    float*    stat2    = (float*)(ws + off);    off = aln(off + 64 * 4);
    _Float16* u16      = (_Float16*)(ws + off); off = aln(off + (size_t)RR * BB * 32 * 2);
    _Float16* wcap16   = (_Float16*)(ws + off); off = aln(off + (size_t)CC * RR * 16 * 32 * 2);
    _Float16* uhat     = (_Float16*)(ws + off); off = aln(off + (size_t)BB * CC * RR * 16 * 2);
    float*    bij      = (float*)(ws + off);    off = aln(off + (size_t)BB * RR * CC * 4);
    float*    cij      = (float*)(ws + off);    off = aln(off + (size_t)BB * RR * CC * 4);
    float*    vbuf     = (float*)(ws + off);    off = aln(off + (size_t)BB * CC * 16 * 4);
    float*    hidden   = (float*)(ws + off);    off = aln(off + (size_t)BB * 256 * 4);
    float*    hidden2  = (float*)(ws + off);    off = aln(off + (size_t)BB * 256 * 4);
    int*      amax     = (int*)(ws + off);      off = aln(off + BB * 4);
    (void)ws_size; (void)in_sizes; (void)n_in; (void)out_size;

    // weight down-conversion (f32 -> f16)
    {
        int n = CC * RR * 16 * 32;
        f32_to_f16_kernel<<<(n + 255) / 256, 256, 0, stream>>>(W_cap, wcap16, n);
    }
    {
        int n = MGEMM * KGEMM;
        f32_to_f16_kernel<<<(n + 255) / 256, 256, 0, stream>>>(prim_w, wprim16, n);
    }

    // conv1 + BN1 + ReLU + im2col
    conv1_kernel<<<BB * L1L, 256, 0, stream>>>(x, conv1_w, conv1_b, conv_raw);
    stats1_kernel<<<256, 256, 0, stream>>>(conv_raw, stat1);
    im2col_bn1_kernel<<<NGEMM, 256, 0, stream>>>(conv_raw, stat1, bn1_g, bn1_b, hcol);

    // primary caps conv: WMMA GEMM 768x3584x1280
    gemm_prim_kernel<<<dim3(MGEMM / 128, NGEMM / 64), 256, 0, stream>>>(
        wprim16, hcol, prim_b, u_raw);

    // BN2 + squash
    stats2_kernel<<<32, 256, 0, stream>>>(u_raw, stat2);
    bn_squash_kernel<<<(BB * RR) / 8, 256, 0, stream>>>(u_raw, stat2, bn2_g, bn2_b, u16);

    // u_hat einsum: WMMA per route
    uhat_kernel<<<RR, 256, 0, stream>>>(wcap16, u16, uhat);

    // dynamic routing (3 iterations)
    {
        int n = BB * RR * CC;
        zero_f32_kernel<<<(n + 255) / 256, 256, 0, stream>>>(bij, n);
    }
    for (int it = 0; it < 3; ++it) {
        softmax_kernel<<<(BB * RR + 255) / 256, 256, 0, stream>>>(bij, cij);
        route_s_kernel<<<BB * CC, 256, 0, stream>>>(uhat, cij, vbuf);
        if (it < 2)
            route_b_kernel<<<BB * CC, 256, 0, stream>>>(uhat, vbuf, bij);
    }

    // classifier + decoder
    cls1_kernel<<<BB, 256, 0, stream>>>(vbuf, cls_w1, cls_b1, hidden);
    cls2_kernel<<<BB, 64, 0, stream>>>(hidden, cls_w2, cls_b2, dout, amax);
    dec1_kernel<<<BB, 256, 0, stream>>>(vbuf, amax, dec_w1, dec_b1, hidden2);
    dec2_kernel<<<BB, 256, 0, stream>>>(hidden2, dec_w2, dec_b2, dout);
}
